// FullAttenSUMNet_60730837566332
// MI455X (gfx1250) — matrix operand order repaired and checked by reading, not compiled, not run on hardware
//
#include <hip/hip_runtime.h>
#include <hip/hip_bf16.h>

// ---------------------------------------------------------------------------
// FullAttenSUMNet for MI455X (gfx1250): WMMA f16 GEMMs for q/k/v projections
// with TDM (tensor_load_to_lds) weight staging, fused per-pixel softmax
// attention, fused BN+ReLU, pool/unpool/concat.
// ---------------------------------------------------------------------------

typedef __attribute__((ext_vector_type(16))) _Float16 v16h;
typedef __attribute__((ext_vector_type(8)))  _Float16 v8h;
typedef __attribute__((ext_vector_type(8)))  float    v8f;
typedef __attribute__((ext_vector_type(4)))  unsigned int u32x4;
typedef __attribute__((ext_vector_type(8)))  int i32x8;
typedef __attribute__((ext_vector_type(4)))  int i32x4;

#define BN_EPS 1e-5f
#define KCHUNK_MAX 256

// ---------------- conversion: f32 activations [C,HW] -> f16 [HW,Kp] --------
__global__ __launch_bounds__(256) void k_cvt_act(const float* __restrict__ x,
    _Float16* __restrict__ out, int C, int HW, int Kp) {
  int i = blockIdx.x * 256 + threadIdx.x;
  int n = HW * Kp;
  if (i >= n) return;
  int hw = i / Kp, c = i - hw * Kp;
  out[i] = (c < C) ? (_Float16)x[(size_t)c * HW + hw] : (_Float16)0.f;
}

// ---------------- conversion: f32 weights [N,C] -> f16 [N,Kp] --------------
__global__ __launch_bounds__(256) void k_cvt_w(const float* __restrict__ w,
    _Float16* __restrict__ out, int N, int C, int Kp) {
  int i = blockIdx.x * 256 + threadIdx.x;
  int n = N * Kp;
  if (i >= n) return;
  int r = i / Kp, c = i - r * Kp;
  out[i] = (c < C) ? (_Float16)w[(size_t)r * C + c] : (_Float16)0.f;
}

// ---------------- WMMA GEMM: Out[N,M] = A[M,Kp] * W[N,Kp]^T ----------------
// Block = 128 threads (4 waves). Block tile: M=64 x N=32. Wave w computes
// rows [w*16, w*16+16) x all 32 N-columns via two v_wmma_f32_16x16x32_f16.
// B weight panel staged into LDS by the Tensor Data Mover once per K-chunk
// (row padding via the TDM pad feature to avoid LDS bank conflicts);
// A tiles double-buffered through LDS.
__global__ __launch_bounds__(128) void k_gemm_wmma(
    const _Float16* __restrict__ A,   // [M, Kp] pixel-major activations
    const _Float16* __restrict__ W,   // [N, Kp] weights
    float* __restrict__ Out,          // [N, M] channel-major output
    int M, int N, int Kp)
{
  __shared__ __align__(16) _Float16 Ald[2][64][40];           // 80B row stride
  __shared__ __align__(16) _Float16 Bpan[32 * (KCHUNK_MAX + 8)];
  const int tid  = threadIdx.x;
  const int wave = tid >> 5;
  const int lane = tid & 31;
  const int tm = blockIdx.x * 64;
  const int tn = blockIdx.y * 32;
  const int row = lane & 15;
  const int hi  = lane >> 4;

  v8f acc0 = {}, acc1 = {};

  // Stage one 64x32 A tile into LDS buffer `buf` (16 halves per thread).
  auto stageA = [&](int buf, int k0) {
    const int r  = tid >> 1;          // 0..63
    const int ko = (tid & 1) * 16;    // 0 or 16 halves
    const _Float16* src = A + (size_t)(tm + r) * Kp + k0 + ko;
    *(v8h*)&Ald[buf][r][ko]     = *(const v8h*)src;
    *(v8h*)&Ald[buf][r][ko + 8] = *(const v8h*)(src + 8);
  };

  for (int kc = 0; kc < Kp; kc += KCHUNK_MAX) {
    const int kchunk  = (Kp - kc < KCHUNK_MAX) ? (Kp - kc) : KCHUNK_MAX;
    const int bstride = kchunk + 8;   // TDM pads 8 halves per row in LDS

#if __has_builtin(__builtin_amdgcn_tensor_load_to_lds)
    if (wave == 0) {
      // ---- Tensor DMA descriptor: 2D tile [32 rows x kchunk halves] ----
      const unsigned long long ga =
          (unsigned long long)(const void*)(W + (size_t)tn * Kp + kc);
      const unsigned ldsa = (unsigned)(size_t)(void*)Bpan;
      const int nd1 = N - tn;                     // rows left (OOB rows -> 0)
      const int pic = (kchunk == 32) ? 3 : (kchunk == 64) ? 4
                    : (kchunk == 128) ? 5 : 6;    // pad every row of DWORDs
      u32x4 g0;
      g0[0] = 1u;                                          // count = 1 (valid)
      g0[1] = ldsa;                                        // lds_addr
      g0[2] = (unsigned)(ga & 0xffffffffu);                // global_addr lo
      g0[3] = (unsigned)((ga >> 32) & 0x01ffffffu) | 0x80000000u;  // type=2
      i32x8 g1;
      g1[0] = (1 << 16) | (1 << 20) | (pic << 22) | (3 << 25);
              // data_size=2B | pad_enable | pad_interval | pad_amount=4 DW
      g1[1] = (int)(((unsigned)Kp & 0xffffu) << 16);       // tensor_dim0 lo
      g1[2] = (int)((((unsigned)Kp >> 16) & 0xffffu) |
                    (((unsigned)nd1 & 0xffffu) << 16));    // dim0 hi|dim1 lo
      g1[3] = (int)((((unsigned)nd1 >> 16) & 0xffffu) |
                    ((unsigned)kchunk << 16));             // dim1 hi|tile_dim0
      g1[4] = 32;                                          // tile_dim1 = 32
      g1[5] = (int)Kp;                                     // dim0_stride lo
      g1[6] = 0;
      g1[7] = 0;
      i32x4 g2 = {0, 0, 0, 0}, g3 = {0, 0, 0, 0};
      i32x8 g4 = {0, 0, 0, 0, 0, 0, 0, 0};
      __builtin_amdgcn_tensor_load_to_lds(g0, g1, g2, g3, g4, 0);
      __builtin_amdgcn_s_wait_tensorcnt(0);
    }
#else
    // Fallback: cooperative staging of the weight panel.
    for (int i = tid; i < 32 * kchunk; i += 128) {
      int nr = i / kchunk, k = i - nr * kchunk;
      Bpan[nr * bstride + k] = (tn + nr < N)
          ? W[(size_t)(tn + nr) * Kp + kc + k] : (_Float16)0.f;
    }
#endif
    __syncthreads();

    // ---- K loop over chunk with double-buffered A staging ----
    stageA(0, kc);
    __syncthreads();
    const int nsteps = kchunk >> 5;
    for (int s = 0; s < nsteps; ++s) {
      const int cur = s & 1;
      if (s + 1 < nsteps) {
        stageA(cur ^ 1, kc + (s + 1) * 32);
        __builtin_prefetch(A + (size_t)(tm + (tid >> 1)) * Kp + kc +
                               (s + 2) * 32, 0, 3);  // global_prefetch_b8
      }
      // A fragment: K contiguous, k = 16*hi + s (matches B fragments).
      const _Float16* ar = &Ald[cur][wave * 16 + row][0];
      v8h a0 = *(const v8h*)(ar + hi * 16);
      v8h a1 = *(const v8h*)(ar + hi * 16 + 8);
      v16h a = __builtin_shufflevector(a0, a1, 0, 1, 2, 3, 4, 5, 6, 7,
                                       8, 9, 10, 11, 12, 13, 14, 15);
      // B fragments: N = lane%16, K contiguous -> straight b128 row reads.
      const int kb = s * 32 + hi * 16;
      const _Float16* b0r = &Bpan[row * bstride + kb];
      const _Float16* b1r = &Bpan[(16 + row) * bstride + kb];
      v8h b00 = *(const v8h*)b0r, b01 = *(const v8h*)(b0r + 8);
      v8h b10 = *(const v8h*)b1r, b11 = *(const v8h*)(b1r + 8);
      v16h b0 = __builtin_shufflevector(b00, b01, 0, 1, 2, 3, 4, 5, 6, 7,
                                        8, 9, 10, 11, 12, 13, 14, 15);
      v16h b1 = __builtin_shufflevector(b10, b11, 0, 1, 2, 3, 4, 5, 6, 7,
                                        8, 9, 10, 11, 12, 13, 14, 15);
      acc0 = __builtin_amdgcn_wmma_f32_16x16x32_f16(
          false, a, false, b0, (short)0, acc0, false, false);
      acc1 = __builtin_amdgcn_wmma_f32_16x16x32_f16(
          false, a, false, b1, (short)0, acc1, false, false);
      __syncthreads();
    }
  }

  // C/D layout: VGPR r -> M = r + 8*hi, N = lane%16.
  const int m0 = tm + wave * 16 + hi * 8;
  const int n0 = tn + row;
  const int n1 = tn + 16 + row;
  if (n0 < N) {
    #pragma unroll
    for (int r = 0; r < 8; ++r) Out[(size_t)n0 * M + m0 + r] = acc0[r];
  }
  if (n1 < N) {
    #pragma unroll
    for (int r = 0; r < 8; ++r) Out[(size_t)n1 * M + m0 + r] = acc1[r];
  }
}

// ---------------- per-pixel 3x3 softmax attention --------------------------
__global__ __launch_bounds__(256) void k_attn(
    const float* __restrict__ q, const float* __restrict__ kbuf,
    const float* __restrict__ vbuf,
    const float* __restrict__ rel_h, const float* __restrict__ rel_w,
    float* __restrict__ out, int O, int H, int W, int relu)
{
  const int p = blockIdx.x;
  const int h = p / W, w = p - h * W;
  const int HW = H * W;
  const int tid = threadIdx.x;
  __shared__ float logits[9];
  if (tid < 9) logits[tid] = 0.f;
  __syncthreads();

  float part[9];
  #pragma unroll
  for (int j = 0; j < 9; ++j) part[j] = 0.f;
  const int Oh = O >> 1;
  for (int o = tid; o < O; o += 256) {
    float qv = q[(size_t)o * HW + p];
    bool isH = o < Oh;
    const float* rel = isH ? (rel_h + (size_t)o * 3)
                           : (rel_w + (size_t)(o - Oh) * 3);
    #pragma unroll
    for (int j = 0; j < 9; ++j) {
      int di = j / 3 - 1, dj = j % 3 - 1;
      int hh = h + di, ww = w + dj;
      float kv = 0.f;  // zero-padded projection outside the image
      if (hh >= 0 && hh < H && ww >= 0 && ww < W)
        kv = kbuf[(size_t)o * HW + hh * W + ww];
      float r = isH ? rel[j / 3] : rel[j % 3];
      part[j] += qv * (kv + r);
    }
  }
  #pragma unroll
  for (int j = 0; j < 9; ++j) atomicAdd(&logits[j], part[j]);  // ds_add_f32
  __syncthreads();

  float mx = logits[0];
  #pragma unroll
  for (int j = 1; j < 9; ++j) mx = fmaxf(mx, logits[j]);
  float e[9]; float s = 0.f;
  #pragma unroll
  for (int j = 0; j < 9; ++j) { e[j] = __expf(logits[j] - mx); s += e[j]; }
  float inv = 1.f / s;

  for (int o = tid; o < O; o += 256) {
    float acc = 0.f;
    #pragma unroll
    for (int j = 0; j < 9; ++j) {
      int di = j / 3 - 1, dj = j % 3 - 1;
      int hh = h + di, ww = w + dj;
      float vv = 0.f;
      if (hh >= 0 && hh < H && ww >= 0 && ww < W)
        vv = vbuf[(size_t)o * HW + hh * W + ww];
      acc += e[j] * inv * vv;
    }
    if (relu) acc = fmaxf(acc, 0.f);
    out[(size_t)o * HW + p] = acc;
  }
}

// ---------------- batchnorm: per-channel stats then apply+relu -------------
__global__ __launch_bounds__(256) void k_bn_stats(const float* __restrict__ x,
    float* __restrict__ mean, float* __restrict__ rstd, int HW) {
  __shared__ float ssum, ssq;
  const int c = blockIdx.x, tid = threadIdx.x;
  if (tid == 0) { ssum = 0.f; ssq = 0.f; }
  __syncthreads();
  float s = 0.f, qq = 0.f;
  const float* xc = x + (size_t)c * HW;
  for (int i = tid; i < HW; i += 256) { float v = xc[i]; s += v; qq += v * v; }
  atomicAdd(&ssum, s);
  atomicAdd(&ssq, qq);
  __syncthreads();
  if (tid == 0) {
    float m = ssum / (float)HW;
    float var = ssq / (float)HW - m * m;
    mean[c] = m;
    rstd[c] = rsqrtf(var + BN_EPS);
  }
}

__global__ __launch_bounds__(256) void k_bn_relu(const float* __restrict__ x,
    const float* __restrict__ mean, const float* __restrict__ rstd,
    const float* __restrict__ gamma, const float* __restrict__ beta,
    float* __restrict__ y, int HW, int n) {
  int i = blockIdx.x * 256 + threadIdx.x;
  if (i >= n) return;
  int c = i / HW;
  float v = gamma[c] * (x[i] - mean[c]) * rstd[c] + beta[c];
  y[i] = fmaxf(v, 0.f);
}

// ---------------- 2x2 maxpool with argmax ----------------------------------
__global__ __launch_bounds__(256) void k_pool2(const float* __restrict__ x,
    float* __restrict__ y, int* __restrict__ idx, int C, int H, int W) {
  const int oh2 = H >> 1, ow2 = W >> 1;
  const int n = C * oh2 * ow2;
  int i = blockIdx.x * 256 + threadIdx.x;
  if (i >= n) return;
  int c = i / (oh2 * ow2);
  int r = i - c * oh2 * ow2;
  int oh = r / ow2, ow = r - oh * ow2;
  const float* xc = x + (size_t)c * H * W;
  float best = xc[(size_t)(2 * oh) * W + 2 * ow];
  int bi = 0;
  #pragma unroll
  for (int j = 1; j < 4; ++j) {
    int dh = j >> 1, dw = j & 1;
    float v = xc[(size_t)(2 * oh + dh) * W + 2 * ow + dw];
    if (v > best) { best = v; bi = j; }  // first-max tie-break like argmax
  }
  y[i] = best;
  idx[i] = bi;
}

// ---------------- unpool (scatter via argmax) into concat dst --------------
__global__ __launch_bounds__(256) void k_unpool(const float* __restrict__ x,
    const int* __restrict__ idx, float* __restrict__ dst,
    int C, int h, int w) {
  const int n = C * h * w;
  int i = blockIdx.x * 256 + threadIdx.x;
  if (i >= n) return;
  int c = i / (h * w);
  int r = i - c * h * w;
  int hh = r / w, ww = r - hh * w;
  float v = x[i];
  int b = idx[i];
  size_t base = (size_t)c * (4 * h * w);
  #pragma unroll
  for (int j = 0; j < 4; ++j) {
    int dh = j >> 1, dw = j & 1;
    dst[base + (size_t)(2 * hh + dh) * (2 * w) + (2 * ww + dw)] =
        (j == b) ? v : 0.f;
  }
}

// ===========================================================================
extern "C" void kernel_launch(void* const* d_in, const int* in_sizes, int n_in,
                              void* d_out, int out_size, void* d_ws, size_t ws_size,
                              hipStream_t stream) {
  (void)in_sizes; (void)n_in; (void)out_size; (void)ws_size;
  char* ws = (char*)d_ws;
  size_t off = 0;
  auto alloc = [&](size_t bytes) -> void* {
    void* r = ws + off;
    off = (off + bytes + 255) & ~(size_t)255;
    return r;
  };

  // --- persistent buffers (skips + pool indices + ping/pong + bn stats) ---
  float* c1   = (float*)alloc((size_t)64  * 16384 * 4);
  float* c2   = (float*)alloc((size_t)128 * 16384 * 4);
  float* c3b  = (float*)alloc((size_t)256 * 4096  * 4);
  float* c4b  = (float*)alloc((size_t)512 * 1024  * 4);
  float* c5b  = (float*)alloc((size_t)512 * 256   * 4);
  int*   i1   = (int*)  alloc((size_t)128 * 4096  * 4);
  int*   i2   = (int*)  alloc((size_t)256 * 1024  * 4);
  int*   i3   = (int*)  alloc((size_t)512 * 256   * 4);
  int*   i4   = (int*)  alloc((size_t)512 * 64    * 4);
  float* bufA = (float*)alloc((size_t)256 * 16384 * 4);
  float* bufB = (float*)alloc((size_t)256 * 16384 * 4);
  float* mean = (float*)alloc(1024 * 4);
  float* rstd = (float*)alloc(1024 * 4);
  const size_t transient_base = off;

  // --- layer helpers ------------------------------------------------------
  // d_in layout: [0]=x, then per layer li: wq,wk,wv,rel_h,rel_w at
  // 1+5*li+{0..4}; then per bn bi: gamma,beta at 86+2*bi+{0,1}.
  auto attn_conv = [&](const float* in, float* outp, int li, int C, int O,
                       int H, int W, int relu, int vonly) {
    off = transient_base;
    const int HW = H * W;
    const int Kp = (C + 31) & ~31;
    _Float16* xf  = (_Float16*)alloc((size_t)HW * Kp * 2);
    _Float16* wqf = (_Float16*)alloc((size_t)O * Kp * 2);
    _Float16* wkf = (_Float16*)alloc((size_t)O * Kp * 2);
    _Float16* wvf = (_Float16*)alloc((size_t)O * Kp * 2);
    const float* wq = (const float*)d_in[1 + 5 * li + 0];
    const float* wk = (const float*)d_in[1 + 5 * li + 1];
    const float* wv = (const float*)d_in[1 + 5 * li + 2];
    const float* rh = (const float*)d_in[1 + 5 * li + 3];
    const float* rw = (const float*)d_in[1 + 5 * li + 4];

    const int na = HW * Kp;
    k_cvt_act<<<(na + 255) / 256, 256, 0, stream>>>(in, xf, C, HW, Kp);
    const int nw = O * Kp;
    k_cvt_w<<<(nw + 255) / 256, 256, 0, stream>>>(wv, wvf, O, C, Kp);
    dim3 g(HW / 64, (O + 31) / 32);
    if (vonly) {  // k=1 layer: softmax over 1 elem == 1, out = v-projection
      k_gemm_wmma<<<g, 128, 0, stream>>>(xf, wvf, outp, HW, O, Kp);
      return;
    }
    k_cvt_w<<<(nw + 255) / 256, 256, 0, stream>>>(wq, wqf, O, C, Kp);
    k_cvt_w<<<(nw + 255) / 256, 256, 0, stream>>>(wk, wkf, O, C, Kp);
    float* qb = (float*)alloc((size_t)O * HW * 4);
    float* kb = (float*)alloc((size_t)O * HW * 4);
    float* vb = (float*)alloc((size_t)O * HW * 4);
    k_gemm_wmma<<<g, 128, 0, stream>>>(xf, wqf, qb, HW, O, Kp);
    k_gemm_wmma<<<g, 128, 0, stream>>>(xf, wkf, kb, HW, O, Kp);
    k_gemm_wmma<<<g, 128, 0, stream>>>(xf, wvf, vb, HW, O, Kp);
    k_attn<<<HW, 256, 0, stream>>>(qb, kb, vb, rh, rw, outp, O, H, W, relu);
  };

  auto bn = [&](const float* in, float* outp, int bi, int C, int HW) {
    const float* gamma = (const float*)d_in[86 + 2 * bi + 0];
    const float* beta  = (const float*)d_in[86 + 2 * bi + 1];
    k_bn_stats<<<C, 256, 0, stream>>>(in, mean, rstd, HW);
    const int n = C * HW;
    k_bn_relu<<<(n + 255) / 256, 256, 0, stream>>>(in, mean, rstd, gamma,
                                                   beta, outp, HW, n);
  };

  auto pool = [&](const float* in, float* outp, int* idx, int C, int H, int W) {
    const int n = C * (H / 2) * (W / 2);
    k_pool2<<<(n + 255) / 256, 256, 0, stream>>>(in, outp, idx, C, H, W);
  };

  auto unpool = [&](const float* in, const int* idx, float* dst, int C,
                    int h, int w) {
    const int n = C * h * w;
    k_unpool<<<(n + 255) / 256, 256, 0, stream>>>(in, idx, dst, C, h, w);
  };

  auto d2d = [&](void* dst, const void* src, size_t bytes) {
    (void)hipMemcpyAsync(dst, src, bytes, hipMemcpyDeviceToDevice, stream);
  };

  const float* x = (const float*)d_in[0];  // [3,128,128]

  // ------------------------------ encoder ---------------------------------
  attn_conv(x,    bufA, 0, 3,   64,  128, 128, 0, 0);  bn(bufA, c1,  0, 64,  16384);
  attn_conv(c1,   bufA, 1, 64,  128, 128, 128, 0, 0);  bn(bufA, c2,  1, 128, 16384);
  pool(c2, bufA, i1, 128, 128, 128);                               // p1 128x64x64
  attn_conv(bufA, bufB, 2, 128, 256, 64, 64, 0, 0);    bn(bufB, bufA, 2, 256, 4096);
  attn_conv(bufA, bufB, 3, 256, 256, 64, 64, 0, 0);    bn(bufB, c3b, 3, 256, 4096);
  pool(c3b, bufA, i2, 256, 64, 64);                                // p2 256x32x32
  attn_conv(bufA, bufB, 4, 256, 512, 32, 32, 0, 0);    bn(bufB, bufA, 4, 512, 1024);
  attn_conv(bufA, bufB, 5, 512, 512, 32, 32, 0, 0);    bn(bufB, c4b, 5, 512, 1024);
  pool(c4b, bufA, i3, 512, 32, 32);                                // p3 512x16x16
  attn_conv(bufA, bufB, 6, 512, 512, 16, 16, 0, 0);    bn(bufB, bufA, 6, 512, 256);
  attn_conv(bufA, bufB, 7, 512, 512, 16, 16, 0, 0);    bn(bufB, c5b, 7, 512, 256);
  pool(c5b, bufA, i4, 512, 16, 16);                                // p4 512x8x8

  // ------------------------------ decoder ---------------------------------
  // u4 = [unpool(p4,i4), c5b]  -> 1024 x 16 x 16 in bufB
  unpool(bufA, i4, bufB, 512, 8, 8);
  d2d(bufB + (size_t)512 * 256, c5b, (size_t)512 * 256 * 4);
  attn_conv(bufB, bufA, 8, 1024, 512, 16, 16, 1, 0);   // d5b
  attn_conv(bufA, bufB, 9, 512,  512, 16, 16, 1, 0);   // d5a
  // u3 = [unpool(d5a,i3), c4b] -> 1024 x 32 x 32 in bufA
  unpool(bufB, i3, bufA, 512, 16, 16);
  d2d(bufA + (size_t)512 * 1024, c4b, (size_t)512 * 1024 * 4);
  attn_conv(bufA, bufB, 10, 1024, 512, 32, 32, 1, 0);  // d4b
  attn_conv(bufB, bufA, 11, 512,  256, 32, 32, 1, 0);  // d4a
  // u2 = [unpool(d4a,i2), c3b] -> 512 x 64 x 64 in bufB
  unpool(bufA, i2, bufB, 256, 32, 32);
  d2d(bufB + (size_t)256 * 4096, c3b, (size_t)256 * 4096 * 4);
  attn_conv(bufB, bufA, 12, 512, 256, 64, 64, 1, 0);   // d3b
  attn_conv(bufA, bufB, 13, 256, 128, 64, 64, 1, 0);   // d3a
  // u1 = [unpool(d3a,i1), c2] -> 256 x 128 x 128 in bufA
  unpool(bufB, i1, bufA, 128, 64, 64);
  d2d(bufA + (size_t)128 * 16384, c2, (size_t)128 * 16384 * 4);
  attn_conv(bufA, bufB, 14, 256, 64, 128, 128, 1, 0);  // d2
  // [d2, c1] -> 128 x 128 x 128 in bufA
  d2d(bufA, bufB, (size_t)64 * 16384 * 4);
  d2d(bufA + (size_t)64 * 16384, c1, (size_t)64 * 16384 * 4);
  attn_conv(bufA, bufB, 15, 128, 32, 128, 128, 1, 0);  // d1
  // output: k=1 -> pure v-projection straight into d_out [2,128,128]
  attn_conv(bufB, (float*)d_out, 16, 32, 2, 128, 128, 0, 1);
}
